// BlockCirculantLayer_7799660609610
// MI455X (gfx1250) — compile-verified, matrix operand order checked by reading, not run on hardware
//
#include <hip/hip_runtime.h>

// ---------------------------------------------------------------------------
// Block-circulant layer == dense GEMM with circulant-expanded weights:
//   out[b, 64*i + t] = sum_{j,s} x[b, 64*j + s] * w[i, j, (t - s) mod 64]
//
// Plan (when scratch is big enough):
//   1) prep_x:  x f32 -> f16  (once, not 64x inline)
//   2) prep_w:  expand circulant -> dense f16 W^T [i][n][K]  (32 MB, L2-resident)
//   3) gemm:    pure f16 WMMA GEMM, 16 wmma per wave per K-panel
// Fallback (small ws): single fused kernel with inline expansion.
// ---------------------------------------------------------------------------

typedef __attribute__((ext_vector_type(16))) _Float16 v16h;
typedef __attribute__((ext_vector_type(8)))  _Float16 v8h;
typedef __attribute__((ext_vector_type(4)))  _Float16 v4h;
typedef __attribute__((ext_vector_type(8)))  float    v8f;
typedef __attribute__((ext_vector_type(4)))  float    v4f;

#define IN_F   4096
#define OUT_F  4096
#define BS     64      // circulant block size
#define PBLK   64      // input blocks
#define QBLK   64      // output blocks
#define LPAD   72      // halfs per LDS row: 144 B (16B-aligned, bank-friendly)

#define CAT16(lo8, hi8) __builtin_shufflevector(lo8, hi8, 0,1,2,3,4,5,6,7, \
                                                           8,9,10,11,12,13,14,15)

// ---------------------------- prep kernels ---------------------------------

__global__ __launch_bounds__(256)
void prep_x_f16(const float* __restrict__ x, _Float16* __restrict__ xh)
{
    const int row = blockIdx.x;
    const float*   xr = x  + (size_t)row * IN_F;
    _Float16*      hr = xh + (size_t)row * IN_F;
    const int tid = threadIdx.x;
    #pragma unroll
    for (int q = 0; q < 4; ++q) {
        int c4 = tid + (q << 8);                 // 1024 float4 slots per row
        v4f f = *(const v4f*)(xr + (c4 << 2));
        v4h h;
        h.x = (_Float16)f.x;  h.y = (_Float16)f.y;
        h.z = (_Float16)f.z;  h.w = (_Float16)f.w;
        *(v4h*)(hr + (c4 << 2)) = h;
    }
}

// Wh[i][n][k] = w[i, k>>6, (n - (k&63)) & 63], f16, K-contiguous rows.
__global__ __launch_bounds__(256)
void prep_w_f16(const float* __restrict__ w, _Float16* __restrict__ wh)
{
    size_t idx = ((size_t)blockIdx.x * 256 + threadIdx.x) << 4;  // 16 halfs each
    int i   = (int)(idx >> 18);          // 64*4096 halfs per i
    int rem = (int)(idx & 0x3FFFF);
    int n   = rem >> 12;                 // 4096 halfs per n row
    int k0  = rem & 4095;                // 16-aligned -> single j block
    int j   = k0 >> 6;
    int s0  = k0 & 63;
    const float* wp = w + ((size_t)(i * PBLK + j) << 6);
    v8h t0, t1;
    #pragma unroll
    for (int u = 0; u < 8; ++u) {
        t0[u] = (_Float16)wp[(n - (s0 + u))     & 63];
        t1[u] = (_Float16)wp[(n - (s0 + u + 8)) & 63];
    }
    *(v8h*)(wh + idx)     = t0;
    *(v8h*)(wh + idx + 8) = t1;
}

// ------------------------------ main GEMM ----------------------------------
// 256 threads = 8 wave32s; tile = 256(M) x 64(N = one output block i).
// Per wave: 2 M-subtiles x 4 N-subtiles -> 8 v8f accumulators, 16 wmma / j.

#define TILE_M 256

__global__ __launch_bounds__(256, 2)
void bc_gemm_f16_pre(const _Float16* __restrict__ xh,
                     const _Float16* __restrict__ wh,
                     float* __restrict__ out)
{
    __shared__ __align__(16) _Float16 Xlds[TILE_M][LPAD];
    __shared__ __align__(16) _Float16 Blds[BS][LPAD];

    const int tid  = threadIdx.x;
    const int wave = tid >> 5;
    const int lane = tid & 31;
    const int lo   = lane & 15;
    const int hi   = lane >> 4;

    const int row0 = blockIdx.x * TILE_M;
    const int iblk = blockIdx.y;

    const _Float16* wpanel = wh + (size_t)iblk * BS * IN_F;   // [n][K]

    v8f acc[2][4];
    #pragma unroll
    for (int mm = 0; mm < 2; ++mm)
        #pragma unroll
        for (int ns = 0; ns < 4; ++ns) acc[mm][ns] = (v8f){};

    for (int j = 0; j < PBLK; ++j) {
        __syncthreads();

        // X panel [256 x 64] f16: 2048 b128 chunks, 8 per thread
        const _Float16* xg = xh + (size_t)row0 * IN_F + j * BS;
        #pragma unroll
        for (int v = 0; v < 8; ++v) {
            int e  = tid + (v << 8);
            int r  = e >> 3;
            int c8 = (e & 7) << 3;
            v8h d = *(const v8h*)(xg + (size_t)r * IN_F + c8);
            *(v8h*)&Xlds[r][c8] = d;
        }
        // B panel [64 x 64] f16: 512 b128 chunks, 2 per thread
        #pragma unroll
        for (int v = 0; v < 2; ++v) {
            int e  = tid + (v << 8);
            int n  = e >> 3;
            int c8 = (e & 7) << 3;
            v8h d = *(const v8h*)(wpanel + (size_t)n * IN_F + j * BS + c8);
            *(v8h*)&Blds[n][c8] = d;
        }
        // prefetch next K panel (global_prefetch_b8)
        if (j + 1 < PBLK) {
            __builtin_prefetch(xg + BS + (size_t)tid * IN_F, 0, 3);
            __builtin_prefetch(wpanel + (size_t)(tid >> 2) * IN_F
                               + (j + 1) * BS + ((tid & 3) << 4), 0, 3);
        }
        __syncthreads();

        #pragma unroll
        for (int ks = 0; ks < 2; ++ks) {
            const int kb = ks * 32 + hi * 8;     // half-wave K offset (ISA layout)
            const int m0 = wave * 32 + lo;
            v8h x0 = *(const v8h*)&Xlds[m0][kb];
            v8h x1 = *(const v8h*)&Xlds[m0][kb + 16];
            v8h y0 = *(const v8h*)&Xlds[m0 + 16][kb];
            v8h y1 = *(const v8h*)&Xlds[m0 + 16][kb + 16];
            v16h a0 = CAT16(x0, x1);
            v16h a1 = CAT16(y0, y1);
            #pragma unroll
            for (int ns = 0; ns < 4; ++ns) {
                const int ncol = ns * 16 + lo;
                v8h b0 = *(const v8h*)&Blds[ncol][kb];
                v8h b1 = *(const v8h*)&Blds[ncol][kb + 16];
                v16h b = CAT16(b0, b1);
                acc[0][ns] = __builtin_amdgcn_wmma_f32_16x16x32_f16(
                    false, a0, false, b, (short)0, acc[0][ns], false, false);
                acc[1][ns] = __builtin_amdgcn_wmma_f32_16x16x32_f16(
                    false, a1, false, b, (short)0, acc[1][ns], false, false);
            }
        }
    }

    // D layout: M = vgpr + 8*hi, N = lane&15
    float* og = out + (size_t)(row0 + wave * 32) * OUT_F + iblk * BS;
    #pragma unroll
    for (int mm = 0; mm < 2; ++mm)
        #pragma unroll
        for (int ns = 0; ns < 4; ++ns)
            #pragma unroll
            for (int v = 0; v < 8; ++v) {
                int m = mm * 16 + v + 8 * hi;
                og[(size_t)m * OUT_F + ns * 16 + lo] = acc[mm][ns][v];
            }
}

// --------------------- fallback: fused single kernel -----------------------

__global__ __launch_bounds__(256, 2)
void bc_gemm_f16_fused(const float* __restrict__ x,
                       const float* __restrict__ w,
                       float* __restrict__ out)
{
    __shared__ __align__(16) _Float16 Xlds[128][LPAD];
    __shared__ __align__(16) _Float16 Blds[BS][LPAD];

    const int tid  = threadIdx.x;
    const int wave = tid >> 5;
    const int lane = tid & 31;
    const int lo   = lane & 15;
    const int hi   = lane >> 4;
    const int row0 = blockIdx.x * 128;
    const int iblk = blockIdx.y;

    v8f acc[4];
    #pragma unroll
    for (int ns = 0; ns < 4; ++ns) acc[ns] = (v8f){};

    for (int j = 0; j < PBLK; ++j) {
        __syncthreads();
        const float* xg = x + (size_t)row0 * IN_F + j * BS;
        #pragma unroll
        for (int v = 0; v < 8; ++v) {
            int e = tid + (v << 8);
            int r = e >> 4, c4 = e & 15;
            v4f f = *(const v4f*)(xg + (size_t)r * IN_F + (c4 << 2));
            v4h h;
            h.x = (_Float16)f.x;  h.y = (_Float16)f.y;
            h.z = (_Float16)f.z;  h.w = (_Float16)f.w;
            *(v4h*)&Xlds[r][c4 << 2] = h;
        }
        {
            const float* wp = w + ((size_t)(iblk * PBLK + j) << 6);
            int n  = tid >> 2;
            int kq = (tid & 3) << 4;
            v8h t0, t1;
            #pragma unroll
            for (int u = 0; u < 8; ++u) {
                t0[u] = (_Float16)wp[(n - (kq + u))     & 63];
                t1[u] = (_Float16)wp[(n - (kq + u + 8)) & 63];
            }
            *(v8h*)&Blds[n][kq]     = t0;
            *(v8h*)&Blds[n][kq + 8] = t1;
        }
        __syncthreads();

        const int mrow = wave * 16 + lo;
        #pragma unroll
        for (int ks = 0; ks < 2; ++ks) {
            const int kb = ks * 32 + hi * 8;
            v8h a_lo = *(const v8h*)&Xlds[mrow][kb];
            v8h a_hi = *(const v8h*)&Xlds[mrow][kb + 16];
            v16h a = CAT16(a_lo, a_hi);
            #pragma unroll
            for (int ns = 0; ns < 4; ++ns) {
                const int ncol = ns * 16 + lo;
                v8h b_lo = *(const v8h*)&Blds[ncol][kb];
                v8h b_hi = *(const v8h*)&Blds[ncol][kb + 16];
                v16h b = CAT16(b_lo, b_hi);
                acc[ns] = __builtin_amdgcn_wmma_f32_16x16x32_f16(
                    false, a, false, b, (short)0, acc[ns], false, false);
            }
        }
    }

    float* og = out + (size_t)(row0 + wave * 16) * OUT_F + iblk * BS;
    #pragma unroll
    for (int ns = 0; ns < 4; ++ns)
        #pragma unroll
        for (int v = 0; v < 8; ++v) {
            int m = v + 8 * hi;
            og[(size_t)m * OUT_F + ns * 16 + lo] = acc[ns][v];
        }
}

// ------------------------------ launcher -----------------------------------

extern "C" void kernel_launch(void* const* d_in, const int* in_sizes, int n_in,
                              void* d_out, int out_size, void* d_ws, size_t ws_size,
                              hipStream_t stream) {
    const float* x = (const float*)d_in[0];   // [N, 4096] f32
    const float* w = (const float*)d_in[1];   // [64, 64, 64] f32
    float* out = (float*)d_out;               // [N, 4096] f32

    const int N = in_sizes[0] / IN_F;

    const size_t xh_bytes = (size_t)N * IN_F * sizeof(_Float16);
    const size_t xh_pad   = (xh_bytes + 255) & ~(size_t)255;
    const size_t wh_bytes = (size_t)QBLK * BS * IN_F * sizeof(_Float16); // 32 MB
    const size_t need     = xh_pad + wh_bytes;

    if (ws_size >= need && (N % TILE_M) == 0) {
        _Float16* xh = (_Float16*)d_ws;
        _Float16* wh = (_Float16*)((char*)d_ws + xh_pad);
        prep_x_f16<<<N, 256, 0, stream>>>(x, xh);
        {
            const size_t wh_halfs = (size_t)QBLK * BS * IN_F;   // 16.78M
            const int    blocks   = (int)(wh_halfs / (16 * 256)); // 4096
            prep_w_f16<<<blocks, 256, 0, stream>>>(w, wh);
        }
        dim3 grid(N / TILE_M, QBLK);
        bc_gemm_f16_pre<<<grid, 256, 0, stream>>>(xh, wh, out);
    } else {
        dim3 grid(N / 128, QBLK);
        bc_gemm_f16_fused<<<grid, 256, 0, stream>>>(x, w, out);
    }
}
